// DisMaxLossFirstPart_19207093747729
// MI455X (gfx1250) — matrix operand order repaired and checked by reading, not compiled, not run on hardware
//
#include <hip/hip_runtime.h>
#include <hip/hip_bf16.h>
#include <math.h>

// ---------------------------------------------------------------------------
// DisMax first part on gfx1250:
//   f = normalize(features); p = normalize(prototypes)
//   dot = f @ p^T  (bf16x3 error-corrected WMMA GEMM, fp32 accumulate)
//   iso = |scale| * sqrt(max(2-2dot, 1e-12)) / sqrt(2)
//   out = -(iso + rowmean(iso)) / temperature
//
// GEMM main loop: double-buffered LDS fed by GLOBAL_LOAD_ASYNC_TO_LDS_B128
// (no VGPR staging -> no spills), one barrier per K-step, async copies of
// tile k+1 overlap the 24 v_wmma_f32_16x16x32_bf16 of tile k.
// ---------------------------------------------------------------------------

typedef __attribute__((ext_vector_type(16))) __bf16        v16bf;
typedef __attribute__((ext_vector_type(8)))  float         v8f;
typedef __attribute__((ext_vector_type(8)))  unsigned int  v8u;

#define KD 1024          // inner dimension (fixed by reference)
#define NKSTEP (KD / 32) // 32 K-steps of BK=32
#define LDSTR 20         // dwords per LDS tile row: 16 data + 4 pad (conflict-free)

// ---- CDNA5 async global->LDS copy (ASYNCcnt-tracked, bypasses VGPRs) ------
__device__ __forceinline__ unsigned lds_off32(const void* p) {
  // generic LDS address = {SHARED_BASE[63:32], lds_offset[31:0]}
  return (unsigned)(unsigned long long)p;
}

__device__ __forceinline__ void async_ld16(unsigned lds, unsigned goff, const void* gbase) {
  // LDS dest offset comes from the VDST VGPR; global address is SGPR base plus VGPR offset (GVS mode)
  asm volatile("global_load_async_to_lds_b128 %0, %1, %2"
               :: "v"(lds), "v"(goff), "s"(gbase) : "memory");
}

__device__ __forceinline__ void wait_async0() {
  asm volatile("s_wait_asynccnt 0" ::: "memory");
}

// A-matrix fragment (16x32 bf16), ISA 7.12.2 layout:
// lane M = lane&15, half = lane>>4; element j: K = (j<8?0:16) + half*8 + (j&7)
// -> two 4-dword runs at dword offsets half*4 and half*4+8 of a k-contiguous row.
__device__ __forceinline__ v16bf load_fragA(const unsigned int* t, int row, int half) {
  const unsigned int* base = t + row * LDSTR + half * 4;
  uint4 a = *(const uint4*)(base);
  uint4 b = *(const uint4*)(base + 8);
  v8u r = { a.x, a.y, a.z, a.w, b.x, b.y, b.z, b.w };
  return __builtin_bit_cast(v16bf, r);
}

// B-matrix fragment (32x16 bf16): lane N = lane&15, half = lane>>4;
// element j: K = (j<8?0:16) + half*8 + (j&7) -> two 4-dword runs at dword offsets half*4 and half*4+8.
__device__ __forceinline__ v16bf load_fragB(const unsigned int* t, int row, int half) {
  const unsigned int* base = t + row * LDSTR + half * 4;
  uint4 a = *(const uint4*)(base);
  uint4 b = *(const uint4*)(base + 8);
  v8u r = { a.x, a.y, a.z, a.w, b.x, b.y, b.z, b.w };
  return __builtin_bit_cast(v16bf, r);
}

// ---------------------------------------------------------------------------
// Kernel 1: per-row L2 normalize, split into hi/lo bf16 pair (x ~= hi + lo)
// ---------------------------------------------------------------------------
__global__ __launch_bounds__(256)
void normalize_split_kernel(const float* __restrict__ src, int rows,
                            __bf16* __restrict__ hi, __bf16* __restrict__ lo)
{
  int row = blockIdx.x;
  if (row >= rows) return;
  const float* x = src + (size_t)row * KD;

  float v[4];
  float ss = 0.f;
#pragma unroll
  for (int i = 0; i < 4; ++i) {
    v[i] = x[threadIdx.x + i * 256];
    ss += v[i] * v[i];
  }
#pragma unroll
  for (int off = 16; off > 0; off >>= 1) ss += __shfl_xor(ss, off, 32);

  __shared__ float red[8];
  __shared__ float s_inv;
  if ((threadIdx.x & 31) == 0) red[threadIdx.x >> 5] = ss;
  __syncthreads();
  if (threadIdx.x == 0) {
    float t = 0.f;
#pragma unroll
    for (int i = 0; i < 8; ++i) t += red[i];
    float n = __builtin_sqrtf(t);
    s_inv = 1.f / fmaxf(n, 1e-12f);   // matches F.normalize eps semantics
  }
  __syncthreads();

  float inv = s_inv;
#pragma unroll
  for (int i = 0; i < 4; ++i) {
    float f = v[i] * inv;
    __bf16 h = (__bf16)f;             // round-to-nearest-even
    float  r = f - (float)h;          // residual captured in lo
    size_t o = (size_t)row * KD + threadIdx.x + i * 256;
    hi[o] = h;
    lo[o] = (__bf16)r;
  }
}

// ---------------------------------------------------------------------------
// Kernel 2: bf16x3 WMMA GEMM + iso epilogue, async double-buffered pipeline.
// Block tile 128(M) x 128(N), BK=32, 256 threads = 8 waves (4 along M, 2 along N),
// each wave computes 32x64 = 2x4 tiles of 16x16.
// ---------------------------------------------------------------------------
__global__ __launch_bounds__(256)
void gemm_iso_kernel(const __bf16* __restrict__ Ahi, const __bf16* __restrict__ Alo,
                     const __bf16* __restrict__ Bhi, const __bf16* __restrict__ Blo,
                     const float* __restrict__ dscale,
                     float* __restrict__ out, int Bm, int Cn)
{
  __shared__ unsigned int sAhi[2][128 * LDSTR];
  __shared__ unsigned int sAlo[2][128 * LDSTR];
  __shared__ unsigned int sBhi[2][128 * LDSTR];
  __shared__ unsigned int sBlo[2][128 * LDSTR];

  const int tid  = threadIdx.x;
  const int lane = tid & 31, wave = tid >> 5;
  const int half = lane >> 4, ln = lane & 15;
  const int waveM = wave & 3, waveN = wave >> 2;
  const int bM = blockIdx.y * 128, bN = blockIdx.x * 128;

  // Issue one 128x32 tile set (A hi/lo + B hi/lo) as async b128 copies.
  // 256 threads x 2 slots cover 128 rows x 4 x 16B per array.
  auto issue_tile = [&](int kb, int buf) {
#pragma unroll
    for (int i = 0; i < 2; ++i) {
      int idx = tid + i * 256;
      int row = idx >> 2, c4 = idx & 3;
      unsigned lo  = (unsigned)((row * LDSTR + c4 * 4) * 4);  // LDS byte offset in tile
      unsigned gco = (unsigned)((kb + c4 * 8) * 2);           // global byte col offset
      int ar = bM + row;
      if (ar < Bm) {
        unsigned goff = (unsigned)ar * (KD * 2) + gco;
        async_ld16(lds_off32(&sAhi[buf][0]) + lo, goff, Ahi);
        async_ld16(lds_off32(&sAlo[buf][0]) + lo, goff, Alo);
      }
      int br = bN + row;
      if (br < Cn) {
        unsigned goff = (unsigned)br * (KD * 2) + gco;
        async_ld16(lds_off32(&sBhi[buf][0]) + lo, goff, Bhi);
        async_ld16(lds_off32(&sBlo[buf][0]) + lo, goff, Blo);
      }
      // OOB rows left unloaded: stale LDS only feeds masked-out outputs.
    }
  };

  v8f acc[2][4];
  {
    v8f z = {0.f, 0.f, 0.f, 0.f, 0.f, 0.f, 0.f, 0.f};
#pragma unroll
    for (int mi = 0; mi < 2; ++mi)
#pragma unroll
      for (int ni = 0; ni < 4; ++ni) acc[mi][ni] = z;
  }

  issue_tile(0, 0);                       // prologue: tile 0 -> buf 0

  for (int ks = 0; ks < NKSTEP; ++ks) {
    wait_async0();                        // my tile-ks copies landed in LDS
    __syncthreads();                      // everyone's landed; buf[nxt] free
    const int cur = ks & 1;
    if (ks + 1 < NKSTEP)
      issue_tile((ks + 1) * 32, cur ^ 1); // overlaps WMMAs below

    v16bf ah[2], al[2], bh[4], bl[4];
#pragma unroll
    for (int mi = 0; mi < 2; ++mi) {
      int r = waveM * 32 + mi * 16 + ln;
      ah[mi] = load_fragA(sAhi[cur], r, half);
      al[mi] = load_fragA(sAlo[cur], r, half);
    }
#pragma unroll
    for (int ni = 0; ni < 4; ++ni) {
      int r = waveN * 64 + ni * 16 + ln;
      bh[ni] = load_fragB(sBhi[cur], r, half);
      bl[ni] = load_fragB(sBlo[cur], r, half);
    }

#pragma unroll
    for (int mi = 0; mi < 2; ++mi)
#pragma unroll
      for (int ni = 0; ni < 4; ++ni) {
        acc[mi][ni] = __builtin_amdgcn_wmma_f32_16x16x32_bf16(
            false, ah[mi], false, bh[ni], (short)0, acc[mi][ni], false, false);
        acc[mi][ni] = __builtin_amdgcn_wmma_f32_16x16x32_bf16(
            false, ah[mi], false, bl[ni], (short)0, acc[mi][ni], false, false);
        acc[mi][ni] = __builtin_amdgcn_wmma_f32_16x16x32_bf16(
            false, al[mi], false, bh[ni], (short)0, acc[mi][ni], false, false);
      }
  }

  // Epilogue: iso = |scale| * sqrt(max(2-2dot, eps)) / sqrt(2)
  // C/D 16x16 f32 layout: VGPR r -> M = mi*16 + half*8 + r, N = lane&15.
  const float s   = fabsf(dscale[0]);
  const float RS2 = 0.70710678118654752440f;
#pragma unroll
  for (int mi = 0; mi < 2; ++mi) {
#pragma unroll
    for (int ni = 0; ni < 4; ++ni) {
      int gN = bN + waveN * 64 + ni * 16 + ln;
      if (gN >= Cn) continue;
      int m0 = bM + waveM * 32 + mi * 16 + half * 8;
#pragma unroll
      for (int r = 0; r < 8; ++r) {
        int m = m0 + r;
        if (m < Bm) {
          float dot = acc[mi][ni][r];
          float sq  = fmaxf(2.f - 2.f * dot, 1e-12f);
          out[(size_t)m * Cn + gN] = s * __builtin_sqrtf(sq) * RS2;
        }
      }
    }
  }
}

// ---------------------------------------------------------------------------
// Kernel 3: deterministic per-row mean of iso (one block per row)
// ---------------------------------------------------------------------------
__global__ __launch_bounds__(256)
void row_mean_kernel(const float* __restrict__ iso, float* __restrict__ rowmean, int Cn)
{
  int row = blockIdx.x;
  float ssum = 0.f;
  for (int c = threadIdx.x; c < Cn; c += 256) ssum += iso[(size_t)row * Cn + c];
#pragma unroll
  for (int off = 16; off > 0; off >>= 1) ssum += __shfl_xor(ssum, off, 32);
  __shared__ float red[8];
  if ((threadIdx.x & 31) == 0) red[threadIdx.x >> 5] = ssum;
  __syncthreads();
  if (threadIdx.x == 0) {
    float t = 0.f;
#pragma unroll
    for (int i = 0; i < 8; ++i) t += red[i];
    rowmean[row] = t / (float)Cn;
  }
}

// ---------------------------------------------------------------------------
// Kernel 4: out = -(iso + rowmean) / temperature   (in-place on d_out)
// ---------------------------------------------------------------------------
__global__ __launch_bounds__(256)
void finalize_kernel(float* __restrict__ out, const float* __restrict__ rowmean,
                     const float* __restrict__ temp, int Cn)
{
  int row = blockIdx.y;
  int c = blockIdx.x * 256 + threadIdx.x;
  if (c < Cn) {
    size_t i = (size_t)row * Cn + c;
    out[i] = -(out[i] + rowmean[row]) / temp[0];
  }
}

// ---------------------------------------------------------------------------
extern "C" void kernel_launch(void* const* d_in, const int* in_sizes, int n_in,
                              void* d_out, int out_size, void* d_ws, size_t ws_size,
                              hipStream_t stream)
{
  const float* features   = (const float*)d_in[0];
  const float* prototypes = (const float*)d_in[1];
  const float* dscale     = (const float*)d_in[2];
  const float* temp       = (const float*)d_in[3];
  float* out = (float*)d_out;

  const int Bm = in_sizes[0] / KD;   // 4096
  const int Cn = in_sizes[1] / KD;   // 10000

  // Workspace layout (bf16 hi/lo operand copies + row means): ~57 MB
  __bf16* fa_hi = (__bf16*)d_ws;
  __bf16* fa_lo = fa_hi + (size_t)Bm * KD;
  __bf16* pb_hi = fa_lo + (size_t)Bm * KD;
  __bf16* pb_lo = pb_hi + (size_t)Cn * KD;
  float*  rowmean = (float*)(pb_lo + (size_t)Cn * KD);

  normalize_split_kernel<<<Bm, 256, 0, stream>>>(features,   Bm, fa_hi, fa_lo);
  normalize_split_kernel<<<Cn, 256, 0, stream>>>(prototypes, Cn, pb_hi, pb_lo);

  dim3 g((Cn + 127) / 128, (Bm + 127) / 128);
  gemm_iso_kernel<<<g, 256, 0, stream>>>(fa_hi, fa_lo, pb_hi, pb_lo, dscale, out, Bm, Cn);

  row_mean_kernel<<<Bm, 256, 0, stream>>>(out, rowmean, Cn);

  dim3 gf((Cn + 255) / 256, Bm);
  finalize_kernel<<<gf, 256, 0, stream>>>(out, rowmean, temp, Cn);
}
